// TAGNN_GCE_PLUS_13116830122157
// MI455X (gfx1250) — matrix operand order repaired and checked by reading, not compiled, not run on hardware
//
#include <hip/hip_runtime.h>
#include <hip/hip_bf16.h>
#include <math.h>

#define BB 512
#define LL 50
#define KK 32
#define DD 100
#define VV 100000
#define KPAD 128
#define BSTRIDE 136   // LDS row stride in halfs (128 + 8 pad -> bank-conflict-free)

typedef __attribute__((ext_vector_type(16))) _Float16 v16h;
typedef __attribute__((ext_vector_type(8)))  _Float16 v8h;
typedef __attribute__((ext_vector_type(8)))  float    v8f;
typedef __attribute__((ext_vector_type(4)))  float    v4f;

__device__ __forceinline__ float sigm(float x) { return 1.0f / (1.0f + __expf(-x)); }

// ---------------------------------------------------------------------------
// Stage 1 of colmean(E_tag): 256 blocks each reduce a contiguous row range.
// Thread c (<100) owns column c -> coalesced 400B row reads. Deterministic.
// ---------------------------------------------------------------------------
__global__ __launch_bounds__(128) void emean_partial(const float* __restrict__ E_tag,
                                                     float* __restrict__ partials) {
  const int c = threadIdx.x;
  if (c >= DD) return;
  const int rowsPer = (VV + 255) / 256;  // 391
  int r0 = blockIdx.x * rowsPer;
  int r1 = min(VV, r0 + rowsPer);
  float acc = 0.0f;
  for (int r = r0; r < r1; ++r) acc += E_tag[r * DD + c];
  partials[blockIdx.x * DD + c] = acc;
}

__global__ __launch_bounds__(128) void emean_final(const float* __restrict__ partials,
                                                   float* __restrict__ e_sum) {
  const int c = threadIdx.x;
  if (c >= DD) return;
  float acc = 0.0f;
  for (int p = 0; p < 256; ++p) acc += partials[p * DD + c];
  e_sum[c] = acc;  // sum; divided by V later
}

// ---------------------------------------------------------------------------
// Per-batch session kernel: chain-GGNN stencil + GRU + target attention +
// global encoder + gate. One block per batch row. Emits sess2 as f16 [128-pad].
// ---------------------------------------------------------------------------
__global__ __launch_bounds__(256) void session_kernel(
    const int* __restrict__ seq, const int* __restrict__ lens,
    const int* __restrict__ nei, const float* __restrict__ wei,
    const float* __restrict__ s_vec, const float* __restrict__ E_tag,
    const float* __restrict__ E_ggnn, const float* __restrict__ E_glob,
    const float* __restrict__ W_ih, const float* __restrict__ b_ih,
    const float* __restrict__ W_hh, const float* __restrict__ b_hh,
    const float* __restrict__ Wout_w, const float* __restrict__ Wout_b,
    const float* __restrict__ W_w,  const float* __restrict__ W3_w,
    const float* __restrict__ W3_b, const float* __restrict__ W1_w,
    const float* __restrict__ W1_b, const float* __restrict__ q1_w,
    const float* __restrict__ W2_w, const float* __restrict__ W2_b,
    const float* __restrict__ G1_w, const float* __restrict__ G1_b,
    const float* __restrict__ G2_w, const float* __restrict__ G2_b,
    const float* __restrict__ e_sum, _Float16* __restrict__ sess2) {
  __shared__ float sx[LL][DD];    // x, then h(final), then score matrix
  __shared__ float shp[LL][DD];   // h_pre, then h_nei (32x100)
  __shared__ float temb[DD], wt[DD], hglb[DD], sess[DD], msg[DD], xg[DD], xg2[DD];
  __shared__ float attL[LL], scoreK[KK], sc[2];

  const int b = blockIdx.x, tid = threadIdx.x;
  const int len = lens[b];
  const int target = seq[b * LL + (len - 1)];

  // x = E_ggnn[seq]
  for (int t = tid; t < LL * DD; t += 256) {
    int l = t / DD, d = t % DD;
    sx[l][d] = E_ggnn[seq[b * LL + l] * DD + d];
  }
  __syncthreads();

  // GRU gates per (l, j) with stencil adjacency folded in
  for (int t = tid; t < LL * DD; t += 256) {
    int l = t / DD, j = t % DD;
    float gir = b_ih[j], giz = b_ih[DD + j], gig = b_ih[2 * DD + j];
    float ghr = b_hh[j], ghz = b_hh[DD + j], ghg = b_hh[2 * DD + j];
    float eo = (l < len - 1) ? 1.0f : 0.0f;
    float ei = (l >= 1 && (l - 1) < len - 1) ? 1.0f : 0.0f;
    for (int d = 0; d < DD; ++d) {
      float xv = sx[l][d];
      float ao = xv + ((l + 1 < LL) ? eo * sx[l + 1][d] : 0.0f);
      float ai = xv + ((l >= 1) ? ei * sx[l - 1][d] : 0.0f);
      const float* wo = W_ih + d * (3 * DD);
      const float* wi = W_ih + (DD + d) * (3 * DD);
      const float* wh = W_hh + d * (3 * DD);
      gir += ao * wo[j]          + ai * wi[j];
      giz += ao * wo[DD + j]     + ai * wi[DD + j];
      gig += ao * wo[2 * DD + j] + ai * wi[2 * DD + j];
      ghr += xv * wh[j];
      ghz += xv * wh[DD + j];
      ghg += xv * wh[2 * DD + j];
    }
    float r = sigm(gir + ghr), z = sigm(giz + ghz);
    float n = tanhf(gig + r * ghg);
    shp[l][j] = (1.0f - z) * n + z * sx[l][j];
  }
  __syncthreads();

  // h = h_pre @ Wout + b  -> overwrite sx
  for (int t = tid; t < LL * DD; t += 256) {
    int l = t / DD, j = t % DD;
    float acc = Wout_b[j];
    for (int d = 0; d < DD; ++d) acc += shp[l][d] * Wout_w[d * DD + j];
    sx[l][j] = acc;  // safe: this phase reads shp, writes sx
  }
  if (tid < DD) temb[tid] = E_tag[target * DD + tid];
  __syncthreads();

  // attention logits
  if (tid < DD) {
    float a = 0.0f;
    for (int e = 0; e < DD; ++e) a += temb[e] * W_w[e * DD + tid];
    wt[tid] = a;
  }
  __syncthreads();
  if (tid < LL) {
    float a = 0.0f;
    for (int d = 0; d < DD; ++d) a += sx[tid][d] * wt[d];
    attL[tid] = a;
  }
  __syncthreads();
  if (tid == 0) {
    float mx = -1e30f;
    for (int l = 0; l < LL; ++l) mx = fmaxf(mx, attL[l]);
    float s = 0.0f;
    for (int l = 0; l < LL; ++l) { attL[l] = __expf(attL[l] - mx); s += attL[l]; }
    for (int l = 0; l < LL; ++l) attL[l] /= s;
  }
  __syncthreads();
  if (tid < DD) {
    float a = 0.0f;
    for (int l = 0; l < LL; ++l) {
      float m = (seq[b * LL + l] > 0) ? 1.0f : 0.0f;
      a += sx[l][tid] * attL[l] * m;
    }
    hglb[tid] = a;
  }
  __syncthreads();
  if (tid < DD) {
    float a = W3_b[tid];
    for (int e = 0; e < DD; ++e) {
      a += hglb[e] * W3_w[e * DD + tid];
      a += sx[len - 1][e] * W3_w[(DD + e) * DD + tid];
      a += temb[e] * W3_w[(2 * DD + e) * DD + tid];
    }
    sess[tid] = tanhf(a);
  }
  __syncthreads();

  // global encoder: h_nei -> shp
  for (int t = tid; t < KK * DD; t += 256) {
    int k = t / DD, d = t % DD;
    shp[k][d] = E_glob[nei[b * KK + k] * DD + d];
  }
  __syncthreads();
  for (int t = tid; t < KK * DD; t += 256) {  // score matrix into sx (h dead)
    int k = t / DD, dp = t % DD;
    float a = W1_b[dp] + wei[b * KK + k] * W1_w[DD * DD + dp];
    for (int d = 0; d < DD; ++d)
      a += shp[k][d] * s_vec[b * DD + d] * W1_w[d * DD + dp];
    sx[k][dp] = tanhf(a) * q1_w[dp];
  }
  __syncthreads();
  if (tid < KK) {
    float a = 0.0f;
    for (int dp = 0; dp < DD; ++dp) a += sx[tid][dp];
    scoreK[tid] = a;
  }
  __syncthreads();
  if (tid == 0) {
    float mx = -1e30f;
    for (int k = 0; k < KK; ++k) mx = fmaxf(mx, scoreK[k]);
    float s = 0.0f;
    for (int k = 0; k < KK; ++k) { scoreK[k] = __expf(scoreK[k] - mx); s += scoreK[k]; }
    for (int k = 0; k < KK; ++k) scoreK[k] /= s;
  }
  __syncthreads();
  if (tid < DD) {
    float a = 0.0f;
    for (int k = 0; k < KK; ++k) a += scoreK[k] * shp[k][tid];
    msg[tid] = a;
    xg[tid] = E_glob[target * DD + tid];
  }
  __syncthreads();
  for (int it = 0; it < 2; ++it) {
    if (tid < DD) {
      float a = W2_b[tid];
      for (int e = 0; e < DD; ++e) {
        a += xg[e] * W2_w[e * DD + tid];
        a += msg[e] * W2_w[(DD + e) * DD + tid];
      }
      xg2[tid] = fmaxf(a, 0.0f);
    }
    __syncthreads();
    if (tid < DD) xg[tid] = xg2[tid];
    __syncthreads();
  }
  if (tid < DD) xg[tid] *= (1.0f - (float)(LL - 1) / (float)LL);  // 0.02
  __syncthreads();

  // mean_b = sess . colmean(E_tag)   (algebraic collapse of [B,V] intermediate)
  if (tid == 0) {
    float a = 0.0f;
    for (int d = 0; d < DD; ++d) a += sess[d] * e_sum[d];
    sc[0] = a / (float)VV;
  }
  __syncthreads();
  if (tid < DD) {
    float a = G1_b[tid] + sc[0] * G1_w[tid];
    for (int d = 0; d < DD; ++d) a += xg[d] * G1_w[(1 + d) * DD + tid];
    xg2[tid] = tanhf(a);
  }
  __syncthreads();
  if (tid == 0) {
    float a = G2_b[0];
    for (int d = 0; d < DD; ++d) a += xg2[d] * G2_w[d];
    sc[1] = sigm(a);
  }
  __syncthreads();
  for (int t = tid; t < KPAD; t += 256) {
    float v = (t < DD) ? (sc[1] * xg[t] + (1.0f - sc[1]) * sc[0]) : 0.0f;
    sess2[b * KPAD + t] = (_Float16)v;
  }
}

// ---------------------------------------------------------------------------
// Final GEMM: out[512,100000] = sess2_f16 @ E_tag^T via WMMA f32_16x16x32_f16.
// Block owns a 64-wide N strip: E_tag[n0..n0+64) is converted f32->f16 ONCE
// into LDS (padded K=128, stride 136 halfs). The 8 waves cover all M=512
// (4 M-tiles x 4 N-tiles = 16 accumulators each). B frags come from LDS
// (ds_load_b128), A frags from the 128KB L2-resident sess2. E_tag is read
// exactly once from HBM (40MB); the kernel is bound by the 204.8MB store.
// ---------------------------------------------------------------------------
__global__ __launch_bounds__(256) void logits_wmma(const _Float16* __restrict__ Af,
                                                   const float* __restrict__ E_tag,
                                                   float* __restrict__ out) {
  __shared__ _Float16 Bs[64 * BSTRIDE];  // 17408 B

  const int tid  = threadIdx.x;
  const int lane = tid & 31;
  const int wv   = tid >> 5;             // 0..7 -> M rows [wv*64, wv*64+64)
  const int n0   = blockIdx.x * 64;

  // ---- stage B strip to LDS as f16 (cooperative, coalesced along K) ----
  for (int t = tid; t < 64 * KPAD; t += 256) {
    int col = t / KPAD, k = t % KPAD;
    int cg = n0 + col;
    if (cg >= VV) cg = VV - 1;           // clamp; stores are guarded
    float v = (k < DD) ? E_tag[cg * DD + k] : 0.0f;
    Bs[col * BSTRIDE + k] = (_Float16)v;
  }
  __syncthreads();

  const int mRow  = lane & 15;           // row within 16-tile
  const int kselA = (lane >> 4) * 16;    // A: low/high-K half per half-wave
  const int nLane = lane & 15;
  const int kselB = (lane >> 4) * 16;    // B: K split 0-15 / 16-31 per half-wave

  v8f c[4][4];
#pragma unroll
  for (int mt = 0; mt < 4; ++mt)
#pragma unroll
    for (int nt = 0; nt < 4; ++nt) c[mt][nt] = (v8f){};

#pragma unroll
  for (int k0 = 0; k0 < KPAD; k0 += 32) {
    // B fragments for the 4 N-subtiles from LDS
    v16h bf[4];
#pragma unroll
    for (int nt = 0; nt < 4; ++nt) {
      const _Float16* pb = Bs + (nt * 16 + nLane) * BSTRIDE + k0 + kselB;
      v8h lo = *(const v8h*)pb;
      v8h hi = *(const v8h*)(pb + 8);
#pragma unroll
      for (int i = 0; i < 8; ++i) { bf[nt][i] = lo[i]; bf[nt][8 + i] = hi[i]; }
    }
    // 4 M-subtiles: A from global (L2-hot), 4 wmmas each
#pragma unroll
    for (int mt = 0; mt < 4; ++mt) {
      const int row = wv * 64 + mt * 16 + mRow;
      const _Float16* pa = Af + row * KPAD + k0 + kselA;
      v8h lo = *(const v8h*)pa;
      v8h hi = *(const v8h*)(pa + 8);
      v16h a;
#pragma unroll
      for (int i = 0; i < 8; ++i) { a[i] = lo[i]; a[8 + i] = hi[i]; }
#pragma unroll
      for (int nt = 0; nt < 4; ++nt)
        c[mt][nt] = __builtin_amdgcn_wmma_f32_16x16x32_f16(
            false, a, false, bf[nt], (short)0, c[mt][nt], false, false);
    }
  }

  // C/D layout: lanes 0-15 hold M=r, lanes 16-31 hold M=8+r; N = lane&15
  const int mHalf = (lane >> 4) * 8;
#pragma unroll
  for (int mt = 0; mt < 4; ++mt) {
    const int mLo = wv * 64 + mt * 16 + mHalf;
#pragma unroll
    for (int nt = 0; nt < 4; ++nt) {
      int col = n0 + nt * 16 + nLane;
      if (col >= VV) continue;
#pragma unroll
      for (int r = 0; r < 8; ++r)
        out[(size_t)(mLo + r) * VV + col] = c[mt][nt][r];
    }
  }
}

// ---------------------------------------------------------------------------
extern "C" void kernel_launch(void* const* d_in, const int* in_sizes, int n_in,
                              void* d_out, int out_size, void* d_ws, size_t ws_size,
                              hipStream_t stream) {
  const int*   seq    = (const int*)  d_in[0];
  const int*   lens   = (const int*)  d_in[1];
  const int*   nei    = (const int*)  d_in[2];
  const float* wei    = (const float*)d_in[3];
  const float* s_vec  = (const float*)d_in[4];
  const float* E_tag  = (const float*)d_in[5];
  const float* E_ggnn = (const float*)d_in[6];
  const float* E_glob = (const float*)d_in[7];
  const float* W_ih   = (const float*)d_in[8];
  const float* b_ih   = (const float*)d_in[9];
  const float* W_hh   = (const float*)d_in[10];
  const float* b_hh   = (const float*)d_in[11];
  const float* Wout_w = (const float*)d_in[12];
  const float* Wout_b = (const float*)d_in[13];
  const float* W_w    = (const float*)d_in[14];
  const float* W3_w   = (const float*)d_in[15];
  const float* W3_b   = (const float*)d_in[16];
  const float* W1_w   = (const float*)d_in[17];
  const float* W1_b   = (const float*)d_in[18];
  const float* q1_w   = (const float*)d_in[19];
  const float* W2_w   = (const float*)d_in[20];
  const float* W2_b   = (const float*)d_in[21];
  const float* G1_w   = (const float*)d_in[22];
  const float* G1_b   = (const float*)d_in[23];
  const float* G2_w   = (const float*)d_in[24];
  const float* G2_b   = (const float*)d_in[25];

  // workspace layout (every word written before read; deterministic):
  //   [0, 102400)        : 256 x 100 float partial column sums
  //   [102400, 102800)   : 100 float column sums of E_tag
  //   [102800, 233872)   : sess2 f16 [512][128] (16B aligned)
  float* partials = (float*)d_ws;
  float* e_sum    = partials + 256 * DD;
  _Float16* sess2 = (_Float16*)((char*)d_ws + 102800);

  emean_partial<<<256, 128, 0, stream>>>(E_tag, partials);
  emean_final<<<1, 128, 0, stream>>>(partials, e_sum);
  session_kernel<<<BB, 256, 0, stream>>>(
      seq, lens, nei, wei, s_vec, E_tag, E_ggnn, E_glob, W_ih, b_ih, W_hh, b_hh,
      Wout_w, Wout_b, W_w, W3_w, W3_b, W1_w, W1_b, q1_w, W2_w, W2_b,
      G1_w, G1_b, G2_w, G2_b, e_sum, sess2);

  const int nBlocks = (VV + 63) / 64;    // 1563 N-strip blocks, 8 waves cover M
  logits_wmma<<<nBlocks, 256, 0, stream>>>(sess2, E_tag, (float*)d_out);
}
